// WMSA_67748814127318
// MI455X (gfx1250) — matrix operand order repaired and checked by reading, not compiled, not run on hardware
//
#include <hip/hip_runtime.h>
#include <hip/hip_bf16.h>

typedef __attribute__((ext_vector_type(16))) __bf16 bf16x16;
typedef __attribute__((ext_vector_type(8)))  __bf16 bf16x8;
typedef __attribute__((ext_vector_type(4)))  __bf16 bf16x4;
typedef __attribute__((ext_vector_type(8)))  float  f32x8;
typedef __attribute__((ext_vector_type(4)))  unsigned int u32x4;
typedef __attribute__((ext_vector_type(4)))  int i32x4;

#define NTOK 49
#define NPAD 64
#define CDIM 128
#define NH   4
#define HDIM 32
#define NWIN 64
#define QK_SCALE 0.17677669529663687f   // 32^-0.5

// ---- fragment loaders -------------------------------------------------------
// A-matrix 16x32 bf16: lane l holds row (l&15);
// lanes 0-15: K = 0..15, lanes 16-31: K = 16..31.
// Caller passes p = &row[k0 + (lane>>4)*16]; we fetch p[0..15].
__device__ __forceinline__ bf16x16 fragA(const __bf16* p) {
    union { u32x4 u[2]; bf16x16 v; } r;
    r.u[0] = *(const u32x4*)(p);
    r.u[1] = *(const u32x4*)(p + 8);
    return r.v;
}
// B-matrix 32x16 bf16: lane l holds column (l&15); lanes 0-15 K=0..15,
// lanes 16-31 K=16..31.  Caller passes p = &col[k0 + (lane>>4)*16]; 16 contiguous.
__device__ __forceinline__ bf16x16 fragB(const __bf16* p) {
    union { u32x4 u[2]; bf16x16 v; } r;
    r.u[0] = *(const u32x4*)(p);
    r.u[1] = *(const u32x4*)(p + 8);
    return r.v;
}
__device__ __forceinline__ f32x8 wmma_bf16(bf16x16 a, bf16x16 b, f32x8 c) {
    return __builtin_amdgcn_wmma_f32_16x16x32_bf16(false, a, false, b,
                                                   (short)0, c, false, false);
}

// ---- weight fp32 -> bf16 pre-pass ------------------------------------------
__global__ void cvt_weights(const float* __restrict__ qkv_w,
                            const float* __restrict__ proj_w,
                            __bf16* __restrict__ wq, __bf16* __restrict__ wp) {
    int i = blockIdx.x * blockDim.x + threadIdx.x;
    if (i < 3 * CDIM * CDIM) wq[i] = (__bf16)qkv_w[i];
    if (i < CDIM * CDIM)     wp[i] = (__bf16)proj_w[i];
}

// ---- fused window-attention kernel: one block (8 wave32) per window --------
__global__ __launch_bounds__(256)
void wmsa_fused(const float* __restrict__ x, const float* __restrict__ attn_mask,
                const __bf16* __restrict__ wq, const float* __restrict__ qkv_b,
                const __bf16* __restrict__ wp, const float* __restrict__ proj_b,
                const float* __restrict__ bias_table, const int* __restrict__ rel_idx,
                float* __restrict__ out)
{
    extern __shared__ char smem[];
    __bf16* xs   = (__bf16*)smem;                 // 64x128 (input; later reused as O)
    __bf16* qs   = xs  + NPAD * CDIM;             // 64x128 (q, pre-scaled)
    __bf16* ks   = qs  + NPAD * CDIM;             // 64x128 (k, row-major)
    __bf16* vsT  = ks  + NPAD * CDIM;             // 128x64 (V, channel-major)
    float*  sbuf = (float*)(vsT + CDIM * NPAD);   // 4x64x64 logits (phase1: x stage)
    __bf16* abuf = (__bf16*)(sbuf + NH * NPAD * NPAD); // 4x64x64 probs bf16

    const int tid  = threadIdx.x;
    const int lane = tid & 31;
    const int wv   = tid >> 5;
    const int lcol = lane & 15;
    const int lhi  = lane >> 4;
    const int b    = blockIdx.x;

    // Warm L2 with the bf16 weights (global_prefetch_b8).
    __builtin_prefetch(wq + (size_t)tid * 192, 0, 0);
    __builtin_prefetch(wp + (size_t)tid * 64, 0, 0);

    // ---- Phase 1: x -> LDS bf16, zero-pad rows 49..63 ----
#if defined(__gfx1250__) && __has_builtin(__builtin_amdgcn_global_load_async_to_lds_b128)
    {
        // Async-copy raw fp32 tile into the sbuf region, then convert in LDS.
        float* stage = sbuf;
        const float* src = x + (size_t)b * NTOK * CDIM;
        for (int i = tid; i < NTOK * (CDIM / 4); i += 256) {
            __builtin_amdgcn_global_load_async_to_lds_b128(
                (i32x4*)(src + i * 4), (i32x4*)(stage + i * 4), 0, 0);
        }
#if __has_builtin(__builtin_amdgcn_s_wait_asynccnt)
        __builtin_amdgcn_s_wait_asynccnt(0);
#else
        asm volatile("s_wait_asynccnt 0x0" ::: "memory");
#endif
        __syncthreads();
        for (int i = tid; i < NPAD * (CDIM / 4); i += 256) {
            const int row = i >> 5;
            const int c4  = (i & 31) * 4;
            bf16x4 h;
            if (row < NTOK) {
                const float4 f = *(const float4*)(stage + row * CDIM + c4);
                h[0] = (__bf16)f.x; h[1] = (__bf16)f.y; h[2] = (__bf16)f.z; h[3] = (__bf16)f.w;
            } else {
                h[0] = (__bf16)0.f; h[1] = (__bf16)0.f; h[2] = (__bf16)0.f; h[3] = (__bf16)0.f;
            }
            *(bf16x4*)(xs + row * CDIM + c4) = h;
        }
    }
#else
    for (int i = tid; i < NPAD * (CDIM / 4); i += 256) {
        const int row = i >> 5;
        const int c4  = (i & 31) * 4;
        bf16x4 h;
        if (row < NTOK) {
            const float4 f = *(const float4*)(x + ((size_t)b * NTOK + row) * CDIM + c4);
            h[0] = (__bf16)f.x; h[1] = (__bf16)f.y; h[2] = (__bf16)f.z; h[3] = (__bf16)f.w;
        } else {
            h[0] = (__bf16)0.f; h[1] = (__bf16)0.f; h[2] = (__bf16)0.f; h[3] = (__bf16)0.f;
        }
        *(bf16x4*)(xs + row * CDIM + c4) = h;
    }
#endif
    __syncthreads();

    // ---- Phase 2: QKV = x @ qkv_w^T + b (64x384) ----
    // Each wave owns 3 column tiles (nt = wv, wv+8, wv+16); B fragments are
    // loaded once per column tile and reused across the 4 M tiles.
    #pragma unroll
    for (int jn = 0; jn < 3; ++jn) {
        const int nt = wv + jn * 8;           // 0..7 -> q, 8..15 -> k, 16..23 -> v
        const int n0 = nt * 16;
        bf16x16 B[4];
        #pragma unroll
        for (int k = 0; k < 4; ++k)
            B[k] = fragB(wq + (size_t)(n0 + lcol) * CDIM + k * 32 + lhi * 16);
        const float bias = qkv_b[n0 + lcol];

        #pragma unroll
        for (int mt = 0; mt < 4; ++mt) {
            f32x8 acc = {};
            #pragma unroll
            for (int k = 0; k < 4; ++k) {
                bf16x16 a = fragA(xs + (mt * 16 + lcol) * CDIM + k * 32 + lhi * 16);
                acc = wmma_bf16(a, B[k], acc);
            }
            if (jn == 0) {            // q tile: scale, row-major (wave-uniform branch)
                #pragma unroll
                for (int j = 0; j < 8; ++j)
                    qs[(mt * 16 + j + lhi * 8) * CDIM + n0 + lcol] =
                        (__bf16)((acc[j] + bias) * QK_SCALE);
            } else if (jn == 1) {     // k tile: row-major
                #pragma unroll
                for (int j = 0; j < 8; ++j)
                    ks[(mt * 16 + j + lhi * 8) * CDIM + (n0 - CDIM) + lcol] =
                        (__bf16)(acc[j] + bias);
            } else {                  // v tile: channel-major -> one packed b128 store
                bf16x8 pk;
                #pragma unroll
                for (int j = 0; j < 8; ++j) pk[j] = (__bf16)(acc[j] + bias);
                *(bf16x8*)(vsT + (n0 - 2 * CDIM + lcol) * NPAD + mt * 16 + lhi * 8) = pk;
            }
        }
    }
    __syncthreads();

    // ---- Phase 3: S = q @ k^T per head; hd=32 -> one WMMA per tile ----
    // wave -> fixed key tile (ntk) and an M-tile pair; K fragment reused.
    {
        const int ntk    = wv & 3;
        const int mtbase = (wv >> 2) * 2;
        #pragma unroll
        for (int h = 0; h < NH; ++h) {
            bf16x16 bb = fragB(ks + (ntk * 16 + lcol) * CDIM + h * HDIM + lhi * 16);
            #pragma unroll
            for (int mi = 0; mi < 2; ++mi) {
                const int m0 = (mtbase + mi) * 16;
                bf16x16 a = fragA(qs + (m0 + lcol) * CDIM + h * HDIM + lhi * 16);
                f32x8 acc = {};
                acc = wmma_bf16(a, bb, acc);
                float* sdst = sbuf + h * NPAD * NPAD + ntk * 16 + lcol;
                #pragma unroll
                for (int j = 0; j < 8; ++j)
                    sdst[(m0 + j + lhi * 8) * NPAD] = acc[j];
            }
        }
    }
    __syncthreads();

    // ---- Phase 4: softmax rows (+rel-pos bias, +mask), write bf16 probs ----
    const int wmask = b & (NWIN - 1);
    const float* mrow0 = attn_mask + (size_t)wmask * NTOK * NTOK;
    for (int r = tid; r < NH * NPAD; r += 256) {
        const int h   = r >> 6;
        const int row = r & 63;
        __bf16* arow = abuf + h * NPAD * NPAD + row * NPAD;
        if (row >= NTOK) {
            for (int c = 0; c < NPAD; ++c) arow[c] = (__bf16)0.f;
            continue;
        }
        float*       srow = sbuf + h * NPAD * NPAD + row * NPAD;
        const float* mrow = mrow0 + row * NTOK;
        const int*   irow = rel_idx + row * NTOK;
        float mx = -1e30f;
        for (int c = 0; c < NTOK; ++c) {
            float v = srow[c] + bias_table[irow[c] * NH + h] + mrow[c];
            srow[c] = v;
            mx = fmaxf(mx, v);
        }
        float sum = 0.f;
        for (int c = 0; c < NTOK; ++c) {
            float e = __expf(srow[c] - mx);
            srow[c] = e;
            sum += e;
        }
        const float inv = 1.f / sum;
        for (int c = 0; c < NTOK; ++c)       arow[c] = (__bf16)(srow[c] * inv);
        for (int c = NTOK; c < NPAD; ++c)    arow[c] = (__bf16)0.f;
    }
    __syncthreads();

    // ---- Phase 5: O = A @ V (per head, 64x32) -> xs (reused) ----
    // wave -> fixed (head, column tile); V fragments reused across M tiles.
    {
        const int h  = wv >> 1;
        const int d0 = (wv & 1) * 16;
        bf16x16 B0 = fragB(vsT + (h * HDIM + d0 + lcol) * NPAD + 0  + lhi * 16);
        bf16x16 B1 = fragB(vsT + (h * HDIM + d0 + lcol) * NPAD + 32 + lhi * 16);
        #pragma unroll
        for (int mt = 0; mt < 4; ++mt) {
            const __bf16* abase = abuf + h * NPAD * NPAD + (mt * 16 + lcol) * NPAD;
            f32x8 acc = {};
            acc = wmma_bf16(fragA(abase + 0  + lhi * 16), B0, acc);
            acc = wmma_bf16(fragA(abase + 32 + lhi * 16), B1, acc);
            #pragma unroll
            for (int j = 0; j < 8; ++j)
                xs[(mt * 16 + j + lhi * 8) * CDIM + h * HDIM + d0 + lcol] = (__bf16)acc[j];
        }
    }
    __syncthreads();

    // ---- Phase 6: Y = O @ proj_w^T + proj_b; wave owns one column tile ----
    {
        const int n0 = wv * 16;
        bf16x16 B[4];
        #pragma unroll
        for (int k = 0; k < 4; ++k)
            B[k] = fragB(wp + (size_t)(n0 + lcol) * CDIM + k * 32 + lhi * 16);
        const float bias = proj_b[n0 + lcol];
        #pragma unroll
        for (int mt = 0; mt < 4; ++mt) {
            f32x8 acc = {};
            #pragma unroll
            for (int k = 0; k < 4; ++k) {
                bf16x16 a = fragA(xs + (mt * 16 + lcol) * CDIM + k * 32 + lhi * 16);
                acc = wmma_bf16(a, B[k], acc);
            }
            #pragma unroll
            for (int j = 0; j < 8; ++j) {
                const int row = mt * 16 + j + lhi * 8;
                if (row < NTOK)
                    out[((size_t)b * NTOK + row) * CDIM + n0 + lcol] = acc[j] + bias;
            }
        }
    }
}

extern "C" void kernel_launch(void* const* d_in, const int* in_sizes, int n_in,
                              void* d_out, int out_size, void* d_ws, size_t ws_size,
                              hipStream_t stream) {
    const float* x         = (const float*)d_in[0];
    const float* attn_mask = (const float*)d_in[1];
    const float* qkv_w     = (const float*)d_in[2];
    const float* qkv_b     = (const float*)d_in[3];
    const float* proj_w    = (const float*)d_in[4];
    const float* proj_b    = (const float*)d_in[5];
    const float* bias_tab  = (const float*)d_in[6];
    const int*   rel_idx   = (const int*)d_in[7];
    float* out = (float*)d_out;

    __bf16* wq = (__bf16*)d_ws;
    __bf16* wp = (__bf16*)((char*)d_ws + (size_t)3 * CDIM * CDIM * sizeof(__bf16));

    cvt_weights<<<(3 * CDIM * CDIM + 255) / 256, 256, 0, stream>>>(qkv_w, proj_w, wq, wp);

    const size_t smem = (size_t)(4 * NPAD * CDIM) * sizeof(__bf16)   // xs,qs,ks,vsT
                      + (size_t)(NH * NPAD * NPAD) * sizeof(float)   // sbuf
                      + (size_t)(NH * NPAD * NPAD) * sizeof(__bf16); // abuf
    wmsa_fused<<<4096, 256, smem, stream>>>(x, attn_mask, wq, qkv_b, wp, proj_b,
                                            bias_tab, rel_idx, out);
}
